// BiLstmBlock_30485677867152
// MI455X (gfx1250) — matrix operand order repaired and checked by reading, not compiled, not run on hardware
//
#include <hip/hip_runtime.h>
#include <hip/hip_bf16.h>

typedef __attribute__((ext_vector_type(16))) __bf16 v16bf;
typedef __attribute__((ext_vector_type(8)))  __bf16 v8bf;
typedef __attribute__((ext_vector_type(8)))  float  v8f;
typedef __attribute__((ext_vector_type(4)))  unsigned int v4u;
typedef __attribute__((ext_vector_type(8)))  int  v8i;
typedef __attribute__((ext_vector_type(4)))  int  v4i;

// Problem constants (from setup_inputs): B=8, Cin=128, P=64, H=W=128
constexpr int  Bb   = 8;
constexpr int  P    = 64;
constexpr int  CX   = 128;          // input channels to each layer (128 both layers)
constexpr int  CTOT = CX + P;       // 192 concatenated channels (x || h)
constexpr int  CO   = 4 * P;        // 256 gate channels
constexpr int  HH   = 128, WW = 128;
constexpr int  NCHUNK = 9 * (CTOT / 32);            // 54 K-chunks of 32
constexpr size_t HW   = (size_t)HH * WW;            // 16384
constexpr size_t NDIR = (size_t)Bb * P * HW;        // 8,388,608 (one direction of a state)
constexpr size_t SZ   = 2 * NDIR;                   // 16,777,216 (one output section)
constexpr size_t WPKN = (size_t)NCHUNK * CO * 32;   // packed bf16 weights per (layer,dir)

// LDS: input tile (bf16, channel-contiguous) unioned with f32 gate tile,
// plus double-buffered 256x32 bf16 weight chunks fed by the TDM. ~109 KB.
struct alignas(16) SmemT {
    union {
        __bf16 in[3][66][CTOT];     // 76,032 B : rows y-1..y+1, x halo, 192 ch
        float  gates[64][CO];       // 65,536 B : 64 pixels x 256 gate channels
    } u;
    __bf16 w[2][CO][32];            // 2 x 16,384 B : weight chunks (TDM dest)
};

__device__ __forceinline__ float sigm(float z) { return 1.f / (1.f + __expf(-z)); }
__device__ __forceinline__ float tanh_fast(float z) { return 2.f / (1.f + __expf(-2.f * z)) - 1.f; }

// Pack one (layer,dir) weight pair into bf16 chunks matching the LDS layout:
// out[((kc*256 + co)*32) + kk] = W(co, ci0+kk, pos), kc = pos*6 + ci0/32.
__global__ __launch_bounds__(256)
void pack_weights_kernel(const float* __restrict__ Wi,   // (CO, CX, 3, 3)
                         const float* __restrict__ Wh,   // (CO, P, 3, 3)
                         __bf16* __restrict__ out)
{
    const int kc  = blockIdx.x;        // 0..53
    const int co  = threadIdx.x;       // 0..255
    const int pos = kc / 6;
    const int ci0 = (kc % 6) * 32;
    __bf16* dst = out + ((size_t)kc * CO + co) * 32;
    #pragma unroll 8
    for (int kk = 0; kk < 32; ++kk) {
        const int ci = ci0 + kk;
        const float v = (ci < CX) ? Wi[((size_t)co * CX + ci) * 9 + pos]
                                  : Wh[((size_t)co * P + (ci - CX)) * 9 + pos];
        dst[kk] = (__bf16)v;
    }
}

// One ConvLSTM cell step for one direction (fused conv-gates + LSTM pointwise).
// X: (B, CX, H, W) f32   Hin/Cst: (B, P, H, W) f32
// Wpk: packed bf16 weights, 54 chunks of 16 KB
__global__ __launch_bounds__(256)
void bilstm_cell_kernel(const float* __restrict__ X,
                        const float* __restrict__ Hin,
                        const float* __restrict__ Cst,
                        const __bf16* __restrict__ Wpk,
                        float* __restrict__ Hout,
                        float* __restrict__ Cout)
{
    const int xt = blockIdx.x;          // 0..1   (64-pixel x tile)
    const int y  = blockIdx.y;          // 0..127
    const int b  = blockIdx.z;          // 0..7
    const int xbase = xt * 64;
    const int tid = threadIdx.x;

    __shared__ SmemT sm;

    const int wave = tid >> 5;
    const int lane = tid & 31;
    const int wm   = wave & 3;          // pixel group: 16 pixels each
    const int wn   = wave >> 2;         // co half: 128 channels each
    const int l16  = lane & 15;
    const int half = lane >> 4;

    // ---- TDM weight-chunk staging (async DMA global->LDS, double-buffered) ----
    const unsigned ldsWbase = (unsigned)__builtin_offsetof(SmemT, w);
    auto issue_chunk = [&](int kc) {
        if (wave == 0) {
            const unsigned long long ga =
                (unsigned long long)(size_t)Wpk + (unsigned long long)kc * 16384ull;
            // D# group0: count=1 | lds_addr | global_addr[56:0] | type=2
            v4u g0 = { 1u,
                       ldsWbase + (unsigned)(kc & 1) * 16384u,
                       (unsigned)ga,
                       ((unsigned)(ga >> 32) & 0x01FFFFFFu) | (2u << 30) };
            // D# group1: data_size=8B; 1-D tensor/tile of 2048 8-byte elements
            v8i g1 = { (int)(3u << 16),      // wg_mask=0 | data_size=3 (8B)
                       (int)(2048u << 16),   // tensor_dim0 lo16 = 2048
                       (int)(1u << 16),      // tensor_dim0 hi=0 | tensor_dim1 lo16 = 1
                       (int)(2048u << 16),   // tensor_dim1 hi=0 | tile_dim0 = 2048
                       1,                    // tile_dim1 = 1 | tile_dim2 = 0
                       2048,                 // tensor_dim0_stride lo32
                       0, 0 };
            v4i zz = { 0, 0, 0, 0 };
#if __clang_major__ >= 23
            v8i z8 = { 0, 0, 0, 0, 0, 0, 0, 0 };
            __builtin_amdgcn_tensor_load_to_lds(g0, g1, zz, zz, z8, 0);
#else
            __builtin_amdgcn_tensor_load_to_lds(g0, g1, zz, zz, 0);
#endif
        }
    };
    issue_chunk(0);   // overlap first DMA with input staging below

    // ---- Stage input tile (x || h) -> LDS bf16, channel-contiguous ----
    // One (channel,row) line per unit: 64 unconditional contiguous loads + 2 edges.
    for (int line = tid; line < 3 * CTOT; line += 256) {
        const int ci  = line / 3;       // 0..191
        const int row = line % 3;
        const int gy  = y + row - 1;
        __bf16* dst = &sm.u.in[row][0][ci];     // xc stride = CTOT elements
        if (gy < 0 || gy >= HH) {
            #pragma unroll 6
            for (int xc = 0; xc < 66; ++xc) dst[(size_t)xc * CTOT] = (__bf16)0.f;
        } else {
            const float* src = (ci < CX)
                ? (X   + (((size_t)b * CX + ci)        * HH + gy) * WW)
                : (Hin + (((size_t)b * P  + (ci - CX)) * HH + gy) * WW);
            dst[0]                     = (__bf16)((xbase - 1 >= 0) ? src[xbase - 1]  : 0.f);
            dst[(size_t)65 * CTOT]     = (__bf16)((xbase + 64 < WW) ? src[xbase + 64] : 0.f);
            #pragma unroll 8
            for (int xc = 1; xc <= 64; ++xc)
                dst[(size_t)xc * CTOT] = (__bf16)src[xbase - 1 + xc];
        }
    }

    v8f acc[8];
    #pragma unroll
    for (int t = 0; t < 8; ++t)
        acc[t] = (v8f){0.f, 0.f, 0.f, 0.f, 0.f, 0.f, 0.f, 0.f};

    // ---- Implicit GEMM: M=64 pixels, N=256 co, K = 9 positions * 192 ch ----
    for (int kc = 0; kc < NCHUNK; ++kc) {
        if (kc + 1 < NCHUNK) {
            issue_chunk(kc + 1);                         // prefetch next chunk (TDM)
            if (wave == 0) __builtin_amdgcn_s_wait_tensorcnt(1);   // chunk kc landed
        } else {
            if (wave == 0) __builtin_amdgcn_s_wait_tensorcnt(0);
        }
        __syncthreads();    // chunk kc visible to all waves (and in[] staged, iter 0)

        const int pos = kc / 6;
        const int ci0 = (kc % 6) * 32;
        const int dy  = pos / 3 - 1;
        const int dx  = pos % 3 - 1;

        // A fragment: 16 pixels x 32 K (bf16), per-lane 2x 16B LDS loads
        // lane<16 holds K {0..7,16..23}; lane>=16 holds K {8..15,24..31}
        const __bf16* abase = &sm.u.in[dy + 1][wm * 16 + l16 + dx + 1][ci0];
        const v8bf a0 = *(const v8bf*)(abase + half * 8);
        const v8bf a1 = *(const v8bf*)(abase + 16 + half * 8);
        const v16bf A = __builtin_shufflevector(a0, a1,
            0, 1, 2, 3, 4, 5, 6, 7, 8, 9, 10, 11, 12, 13, 14, 15);

        const __bf16 (*wbuf)[32] = sm.w[kc & 1];
        #pragma unroll
        for (int t = 0; t < 8; ++t) {
            // B fragment: 32 K x 16 co; lane n = co, K {8h..8h+7, 16+8h..16+8h+7}
            const int co = wn * 128 + t * 16 + l16;
            const __bf16* bbase = &wbuf[co][half * 8];
            const v8bf b0 = *(const v8bf*)(bbase);
            const v8bf b1 = *(const v8bf*)(bbase + 16);
            const v16bf Bf = __builtin_shufflevector(b0, b1,
                0, 1, 2, 3, 4, 5, 6, 7, 8, 9, 10, 11, 12, 13, 14, 15);
            acc[t] = __builtin_amdgcn_wmma_f32_16x16x32_bf16(
                false, A, false, Bf, (short)0, acc[t], false, false);
        }
        __syncthreads();    // all waves done with w[kc&1]; next-next DMA may overwrite
    }

    // ---- Spill gate accumulators to LDS (reuses input-stage storage) ----
    #pragma unroll
    for (int t = 0; t < 8; ++t) {
        const int co = wn * 128 + t * 16 + l16;
        #pragma unroll
        for (int r = 0; r < 8; ++r) {
            const int px = wm * 16 + r + 8 * half;  // C/D layout: M = r + 8*(lane/16)
            sm.u.gates[px][co] = acc[t][r];
        }
    }
    __syncthreads();

    // ---- LSTM pointwise: i,f,g,o -> c_new, h_new ----
    #pragma unroll
    for (int j = 0; j < 16; ++j) {
        const int idx = tid + 256 * j;      // 0..4095 over (p, px)
        const int px  = idx & 63;           // x within tile (coalesced stores)
        const int p   = idx >> 6;           // hidden channel 0..63
        const float gi = sm.u.gates[px][p];
        const float gf = sm.u.gates[px][64 + p];
        const float gg = sm.u.gates[px][128 + p];
        const float go = sm.u.gates[px][192 + p];
        const size_t off = (((size_t)b * P + p) * HH + y) * WW + (xbase + px);
        const float c_old = Cst[off];
        const float cn = sigm(gf) * c_old + sigm(gi) * tanh_fast(gg);
        const float hn = sigm(go) * tanh_fast(cn);
        Cout[off] = cn;
        Hout[off] = hn;
    }
}

// Batch-norm statistics over (B, H, W) per channel of concat(h_dir0, h_dir1).
// Hcat layout: (2, B, P, H, W); logical channel c: dir=c/64, p=c%64.
__global__ __launch_bounds__(256)
void bn_stats_kernel(const float* __restrict__ Hcat,
                     float* __restrict__ mean, float* __restrict__ rstd)
{
    const int c   = blockIdx.x;     // 0..127
    const int dir = c >> 6;
    const int p   = c & 63;
    float s = 0.f, s2 = 0.f;
    for (int b = 0; b < Bb; ++b) {
        const float* sb = Hcat + ((((size_t)dir * Bb + b) * P + p) * HW);
        for (int i = threadIdx.x; i < (int)HW; i += 256) {
            const float v = sb[i];
            s += v; s2 += v * v;
        }
    }
    __shared__ float rs[256], rq[256];
    rs[threadIdx.x] = s; rq[threadIdx.x] = s2;
    __syncthreads();
    for (int st = 128; st > 0; st >>= 1) {
        if (threadIdx.x < st) {
            rs[threadIdx.x] += rs[threadIdx.x + st];
            rq[threadIdx.x] += rq[threadIdx.x + st];
        }
        __syncthreads();
    }
    if (threadIdx.x == 0) {
        const float n = (float)(Bb * HW);
        const float m = rs[0] / n;
        const float v = rq[0] / n - m * m;
        mean[c] = m;
        rstd[c] = rsqrtf(v + 1e-5f);
    }
}

// out1 = relu(bn(concat(h))), written as (B, 2P, H, W)
__global__ __launch_bounds__(256)
void bn_relu_kernel(const float* __restrict__ Hcat,
                    const float* __restrict__ gamma, const float* __restrict__ beta,
                    const float* __restrict__ mean,  const float* __restrict__ rstd,
                    float* __restrict__ Out)
{
    const size_t i = (size_t)blockIdx.x * 256 + threadIdx.x;   // over SZ
    const int    hw = (int)(i & (HW - 1));
    const size_t r  = i >> 14;
    const int    c  = (int)(r & 127);
    const int    b  = (int)(r >> 7);
    const float v = Hcat[((((size_t)(c >> 6) * Bb + b) * P) + (c & 63)) * HW + hw];
    const float o = gamma[c] * (v - mean[c]) * rstd[c] + beta[c];
    Out[i] = fmaxf(o, 0.f);
}

// out = relu(bn(concat(h2)) + identity_x)
__global__ __launch_bounds__(256)
void bn_residual_relu_kernel(const float* __restrict__ Hcat,
                             const float* __restrict__ gamma, const float* __restrict__ beta,
                             const float* __restrict__ mean,  const float* __restrict__ rstd,
                             const float* __restrict__ Xid,
                             float* __restrict__ Out)
{
    const size_t i = (size_t)blockIdx.x * 256 + threadIdx.x;
    const int    hw = (int)(i & (HW - 1));
    const size_t r  = i >> 14;
    const int    c  = (int)(r & 127);
    const int    b  = (int)(r >> 7);
    const float v = Hcat[((((size_t)(c >> 6) * Bb + b) * P) + (c & 63)) * HW + hw];
    const float o = gamma[c] * (v - mean[c]) * rstd[c] + beta[c] + Xid[i];
    Out[i] = fmaxf(o, 0.f);
}

extern "C" void kernel_launch(void* const* d_in, const int* in_sizes, int n_in,
                              void* d_out, int out_size, void* d_ws, size_t ws_size,
                              hipStream_t stream)
{
    (void)in_sizes; (void)n_in; (void)out_size; (void)ws_size;

    const float* x      = (const float*)d_in[0];
    const float* h1_h   = (const float*)d_in[1];
    const float* h1_c   = (const float*)d_in[2];
    const float* h2_h   = (const float*)d_in[3];
    const float* h2_c   = (const float*)d_in[4];
    const float* w_ih1  = (const float*)d_in[5];
    const float* w_hh1  = (const float*)d_in[6];
    const float* w_ih2  = (const float*)d_in[7];
    const float* w_hh2  = (const float*)d_in[8];
    const float* gamma1 = (const float*)d_in[9];
    const float* beta1  = (const float*)d_in[10];
    const float* gamma2 = (const float*)d_in[11];
    const float* beta2  = (const float*)d_in[12];

    float* out = (float*)d_out;                  // (B, 2P, H, W)
    float* h1h = out + SZ;                       // (2, B, P, H, W)
    float* h1c = out + 2 * SZ;
    float* h2h = out + 3 * SZ;
    float* h2c = out + 4 * SZ;

    float* out1  = (float*)d_ws;                 // (B, 2P, H, W) layer-2 input
    float* mean1 = out1 + SZ;
    float* rstd1 = mean1 + 128;
    float* mean2 = rstd1 + 128;
    float* rstd2 = mean2 + 128;
    __bf16* wpk  = (__bf16*)((char*)d_ws + SZ * sizeof(float) + 4096);
    // packed weights: [layer1 d0][layer1 d1][layer2 d0][layer2 d1], 864 KB each

    const dim3 cgrid(2, HH, Bb);                 // 64-px x-tiles, rows, batch

    // ---- Pack weights to bf16 TDM-friendly chunks (once per launch) ----
    for (int d = 0; d < 2; ++d) {
        pack_weights_kernel<<<NCHUNK, 256, 0, stream>>>(
            w_ih1 + (size_t)d * CO * CX * 9, w_hh1 + (size_t)d * CO * P * 9,
            wpk + (size_t)d * WPKN);
        pack_weights_kernel<<<NCHUNK, 256, 0, stream>>>(
            w_ih2 + (size_t)d * CO * CX * 9, w_hh2 + (size_t)d * CO * P * 9,
            wpk + (size_t)(2 + d) * WPKN);
    }

    // ---- Layer 1: two directions, fused conv-gates + LSTM pointwise ----
    for (int d = 0; d < 2; ++d) {
        bilstm_cell_kernel<<<cgrid, 256, 0, stream>>>(
            x,
            h1_h + (size_t)d * NDIR,
            h1_c + (size_t)d * NDIR,
            wpk + (size_t)d * WPKN,
            h1h + (size_t)d * NDIR,
            h1c + (size_t)d * NDIR);
    }
    bn_stats_kernel<<<128, 256, 0, stream>>>(h1h, mean1, rstd1);
    bn_relu_kernel<<<(unsigned)(SZ / 256), 256, 0, stream>>>(
        h1h, gamma1, beta1, mean1, rstd1, out1);

    // ---- Layer 2 ----
    for (int d = 0; d < 2; ++d) {
        bilstm_cell_kernel<<<cgrid, 256, 0, stream>>>(
            out1,
            h2_h + (size_t)d * NDIR,
            h2_c + (size_t)d * NDIR,
            wpk + (size_t)(2 + d) * WPKN,
            h2h + (size_t)d * NDIR,
            h2c + (size_t)d * NDIR);
    }
    bn_stats_kernel<<<128, 256, 0, stream>>>(h2h, mean2, rstd2);
    bn_residual_relu_kernel<<<(unsigned)(SZ / 256), 256, 0, stream>>>(
        h2h, gamma2, beta2, mean2, rstd2, x, out);
}
